// SelfAttention_29618094474217
// MI455X (gfx1250) — matrix operand order repaired and checked by reading, not compiled, not run on hardware
//
#include <hip/hip_runtime.h>
#include <hip/hip_bf16.h>

typedef __attribute__((ext_vector_type(16))) __bf16 v16bf;
typedef __attribute__((ext_vector_type(8)))  float  v8f;

namespace {
constexpr int NPIX = 4096;   // H*W
constexpr int CCH  = 64;     // channels
constexpr int DQK  = 8;      // q/k inner dim
constexpr int WAVES = 8;     // waves per block in attention kernel
constexpr int LDS_STRIDE = 80;            // bytes per 32-wide bf16 P row (conflict-free, 16B aligned)
constexpr int LDS_TILE   = 16 * LDS_STRIDE;
constexpr float CEXP = 0.51012695f;       // (1/sqrt(8)) * log2(e), folded into q
}

union FragAB { v16bf v; uint4 u[2]; };

// ---------------------------------------------------------------------------
// Projection: rows 0-7 -> qT[b][i][8] (bf16, pre-scaled by CEXP so the attention
// kernel can exp2() the raw WMMA score), 8-15 -> kT[b][j][8], 16-79 -> v.
// v is stored j-interleaved within each 32-wide block: storage position
// p = (j%32 < 16) ? 2*(j%32) : 2*(j%32-16)+1, i.e. j = (p&1)*16 + p/2 -- the
// same K-permutation the attention kernel uses when spilling the P tile.
// grid = (N/256, 80, B), block = 256. Weight row is block-uniform (scalar loads).
// ---------------------------------------------------------------------------
__global__ __launch_bounds__(256) void sa_proj_kernel(
    const float* __restrict__ x,
    const float* __restrict__ Wq, const float* __restrict__ bq,
    const float* __restrict__ Wk, const float* __restrict__ bk,
    const float* __restrict__ Wv, const float* __restrict__ bv,
    __hip_bfloat16* __restrict__ qT, __hip_bfloat16* __restrict__ kT,
    __hip_bfloat16* __restrict__ vb) {
  const int i   = blockIdx.x * 256 + threadIdx.x;
  const int row = blockIdx.y;
  const int b   = blockIdx.z;

  const float* wrow;
  float acc;
  if (row < 8)       { wrow = Wq + row * CCH;        acc = bq[row];      }
  else if (row < 16) { wrow = Wk + (row - 8) * CCH;  acc = bk[row - 8];  }
  else               { wrow = Wv + (row - 16) * CCH; acc = bv[row - 16]; }

  const float* xb = x + (size_t)b * CCH * NPIX + i;   // x[b][c][i], coalesced over i
#pragma unroll 8
  for (int c = 0; c < CCH; ++c) acc = fmaf(wrow[c], xb[(size_t)c * NPIX], acc);

  if (row < 8) {
    qT[((size_t)b * NPIX + i) * DQK + row] = __float2bfloat16(acc * CEXP);
  } else if (row < 16) {
    kT[((size_t)b * NPIX + i) * DQK + (row - 8)] = __float2bfloat16(acc);
  } else {
    const int il = i & 31;
    const int p  = (il < 16) ? (2 * il) : (2 * (il - 16) + 1);   // interleave perm
    vb[((size_t)b * CCH + (row - 16)) * NPIX + (i & ~31) + p] = __float2bfloat16(acc);
  }
}

// ---------------------------------------------------------------------------
// Flash-style attention, TWO 16-query tiles per wave (shares k/v fragment loads
// across both P tiles -> 12 WMMAs per 10 global b128 loads per 32-j block).
// Single pass over j with unnormalized accumulation (out_u += exp(s)*v,
// Z += exp(s)), normalized at the end.
// ---------------------------------------------------------------------------
__global__ __launch_bounds__(WAVES * 32) void sa_attn_kernel(
    const __hip_bfloat16* __restrict__ qT, const __hip_bfloat16* __restrict__ kT,
    const __hip_bfloat16* __restrict__ vb, const float* __restrict__ x,
    const float* __restrict__ gamma, float* __restrict__ out) {
  __shared__ char lds[WAVES * 2 * LDS_TILE];

  const int tid  = threadIdx.x;
  const int wave = tid >> 5;
  const int lane = tid & 31;
  const int hl   = lane >> 4;    // lane half (0 or 1)
  const int lrow = lane & 15;

  const int pair = blockIdx.x * WAVES + wave;   // 512 pairs of i-tiles
  const int b    = pair >> 7;                   // 128 pairs per batch
  const int i0   = (pair & 127) << 5;           // 32 queries per wave

  char* my[2];
  my[0] = lds + wave * (2 * LDS_TILE);
  my[1] = my[0] + LDS_TILE;

  const uint4 z4 = make_uint4(0u, 0u, 0u, 0u);

  // A-fragments of q for both i-tiles (16 x 32, K = d padded 8->32 with zeros).
  // Only lanes 0-15, elements 0-7 carry data (16 contiguous bytes per row).
  FragAB aq[2];
#pragma unroll
  for (int u = 0; u < 2; ++u) {
    const uint4 qv = *(const uint4*)(qT + ((size_t)b * NPIX + i0 + 16 * u + lrow) * DQK);
    aq[u].u[0] = (hl == 0) ? qv : z4;
    aq[u].u[1] = z4;
  }

  // Lane base pointer for k fragments: kT[b][jb + lrow][0..7] = kbase[jb] (16B units).
  const uint4* kbase = (const uint4*)(kT + ((size_t)b * NPIX + lrow) * DQK);

  v8f acc[2][4] = {{v8f{}, v8f{}, v8f{}, v8f{}}, {v8f{}, v8f{}, v8f{}, v8f{}}};
  float zacc[2][8] = {{0.f,0.f,0.f,0.f,0.f,0.f,0.f,0.f},
                      {0.f,0.f,0.f,0.f,0.f,0.f,0.f,0.f}};

  for (int jb = 0; jb < NPIX; jb += 32) {
    // k B-fragments for two 16-wide j tiles: col=lane%16 -> j, element e -> K=16*hl+e (K<8 valid)
    FragAB bk0, bk1;
    {
      const uint4 k0 = kbase[jb];
      const uint4 k1 = kbase[jb + 16];
      bk0.u[0] = (hl == 0) ? k0 : z4;  bk0.u[1] = z4;
      bk1.u[0] = (hl == 0) ? k1 : z4;  bk1.u[1] = z4;
    }

    // Scores: 2 i-tiles x 2 j-tiles. q already carries (1/sqrt(8))*log2(e).
    const v8f zc = {};
    v8f sc[2][2];
#pragma unroll
    for (int u = 0; u < 2; ++u) {
      sc[u][0] = __builtin_amdgcn_wmma_f32_16x16x32_bf16(false, aq[u].v, false, bk0.v, (short)0, zc, false, false);
      sc[u][1] = __builtin_amdgcn_wmma_f32_16x16x32_bf16(false, aq[u].v, false, bk1.v, (short)0, zc, false, false);
    }

    // exp2 (raw v_exp_f32), running Z, and spill P tiles to LDS in interleaved K
    // order: position 2*lrow <- tile0, 2*lrow+1 <- tile1 (one packed bf16x2 store).
#pragma unroll
    for (int u = 0; u < 2; ++u) {
#pragma unroll
      for (int r = 0; r < 8; ++r) {
        const float p0 = __builtin_amdgcn_exp2f(sc[u][0][r]);
        const float p1 = __builtin_amdgcn_exp2f(sc[u][1][r]);
        zacc[u][r] += p0 + p1;
        __hip_bfloat162 pk;
        pk.x = __float2bfloat16(p0);
        pk.y = __float2bfloat16(p1);
        *(__hip_bfloat162*)(my[u] + (r + 8 * hl) * LDS_STRIDE + lrow * 4) = pk;
      }
    }
    asm volatile("s_wait_dscnt 0" ::: "memory");   // single wave: order LDS store -> read

    // Re-read P tiles as A-fragments: row=lane%16; element e -> K = 16*hl + e.
    FragAB ap[2];
#pragma unroll
    for (int u = 0; u < 2; ++u) {
      ap[u].u[0] = *(const uint4*)(my[u] + lrow * LDS_STRIDE + 32 * hl);
      ap[u].u[1] = *(const uint4*)(my[u] + lrow * LDS_STRIDE + 32 * hl + 16);
    }

    // out[32 i x 64 c] += P * V^T.  v B-fragment loads are shared by both i-tiles.
    // B-fragment: col c = 16t + lane%16, element e -> K = 16*hl + e; vb is stored in
    // the same interleaved K order, so the 16B loads stay contiguous.
#pragma unroll
    for (int t = 0; t < 4; ++t) {
      const __hip_bfloat16* vp = vb + ((size_t)b * CCH + t * 16 + lrow) * NPIX + jb + 16 * hl;
      FragAB bvf;
      bvf.u[0] = *(const uint4*)(vp);
      bvf.u[1] = *(const uint4*)(vp + 8);
      acc[0][t] = __builtin_amdgcn_wmma_f32_16x16x32_bf16(false, ap[0].v, false, bvf.v, (short)0, acc[0][t], false, false);
      acc[1][t] = __builtin_amdgcn_wmma_f32_16x16x32_bf16(false, ap[1].v, false, bvf.v, (short)0, acc[1][t], false, false);
    }
  }

  // Reduce Z over the 16 lanes of each half (rows r+8*hl live in that half), invert,
  // then fused epilogue: out = gamma * (out_u / Z) + x.
  const float g = gamma[0];
#pragma unroll
  for (int u = 0; u < 2; ++u) {
#pragma unroll
    for (int r = 0; r < 8; ++r) {
      float z = zacc[u][r];
      z += __shfl_xor(z, 1);
      z += __shfl_xor(z, 2);
      z += __shfl_xor(z, 4);
      z += __shfl_xor(z, 8);
      zacc[u][r] = 1.0f / z;
    }
#pragma unroll
    for (int r = 0; r < 8; ++r) {
      const int i = i0 + 16 * u + r + 8 * hl;
      const size_t base = (size_t)b * CCH * NPIX + i;
#pragma unroll
      for (int t = 0; t < 4; ++t) {
        const size_t idx = base + (size_t)(t * 16 + lrow) * NPIX;
        out[idx] = fmaf(g, acc[u][t][r] * zacc[u][r], x[idx]);
      }
    }
  }
}

// ---------------------------------------------------------------------------
extern "C" void kernel_launch(void* const* d_in, const int* in_sizes, int n_in,
                              void* d_out, int out_size, void* d_ws, size_t ws_size,
                              hipStream_t stream) {
  const float* x     = (const float*)d_in[0];
  const float* Wq    = (const float*)d_in[1];
  const float* bq    = (const float*)d_in[2];
  const float* Wk    = (const float*)d_in[3];
  const float* bk    = (const float*)d_in[4];
  const float* Wv    = (const float*)d_in[5];
  const float* bv    = (const float*)d_in[6];
  const float* gamma = (const float*)d_in[7];
  float* out = (float*)d_out;

  char* ws = (char*)d_ws;
  __hip_bfloat16* qT = (__hip_bfloat16*)(ws);                 // 4*4096*8  bf16 = 256 KB
  __hip_bfloat16* kT = (__hip_bfloat16*)(ws + (256 << 10));   // 256 KB
  __hip_bfloat16* vb = (__hip_bfloat16*)(ws + (512 << 10));   // 4*64*4096 bf16 = 2 MB

  dim3 pg(NPIX / 256, DQK + DQK + CCH, 4);   // (16, 80, 4)
  sa_proj_kernel<<<pg, dim3(256), 0, stream>>>(x, Wq, bq, Wk, bk, Wv, bv, qT, kT, vb);

  // 512 pairs of 16-query tiles = 64 blocks * 8 waves
  sa_attn_kernel<<<dim3(64), dim3(WAVES * 32), 0, stream>>>(qT, kT, vb, x, gamma, out);
}
